// Attention_13056700580171
// MI455X (gfx1250) — matrix operand order repaired and checked by reading, not compiled, not run on hardware
//
#include <hip/hip_runtime.h>
#include <stdint.h>

// Problem constants (from reference)
#define B_   128
#define S_   1024
#define EH2_ 1024
#define DH_  512
#define A_   512
// Tiling for the WMMA scores kernel
#define MT   64              // S-rows per block
#define KT   32              // K step (bf16 WMMA K)
#define NKT  (EH2_ / KT)     // 32 K-steps

typedef __attribute__((ext_vector_type(16))) __bf16 v16bf;
typedef __attribute__((ext_vector_type(8)))  float  v8f;
typedef __attribute__((ext_vector_type(4)))  int    v4i;

union ABfrag { v16bf bf; uint4 q[2]; };

// ---- gfx1250 async global->LDS copy (ASYNCcnt) with portable fallback ----
#if defined(__has_builtin)
#  if __has_builtin(__builtin_amdgcn_global_load_async_to_lds_b128)
#    define HAS_ASYNC_LDS 1
#  endif
#endif
#ifndef HAS_ASYNC_LDS
#  define HAS_ASYNC_LDS 0
#endif

#if HAS_ASYNC_LDS
typedef __attribute__((address_space(1))) v4i GV4;   // global 16B chunk
typedef __attribute__((address_space(3))) v4i LV4;   // LDS 16B chunk
static __device__ __forceinline__ void async_wait_all() {
#  if __has_builtin(__builtin_amdgcn_s_wait_asynccnt)
  __builtin_amdgcn_s_wait_asynccnt(0);
#  else
  asm volatile("s_wait_asynccnt 0x0" ::: "memory");
#  endif
}
#else
static __device__ __forceinline__ void async_wait_all() {}
#endif

static __device__ __forceinline__ unsigned short f32_to_bf16(float f) {
  unsigned int u = __float_as_uint(f);
  u += 0x7fffu + ((u >> 16) & 1u);   // round-to-nearest-even
  return (unsigned short)(u >> 16);
}

// ---------------------------------------------------------------------------
// Kernel 1: convert W_enc (f32, [K=1024][N=512] row-major) into bf16, stored in
// WMMA B-fragment order:  index = (((kt*32 + ntg)*2 + h)*32 + lane)*8 + e
// Contiguous k order used here (16-bit B 32x16): lanes 0-15 hold K=0..15 of col N=lane,
// lanes 16-31 hold K=16..31 of col N=lane-16; i = h*8+e selects K within half.
// ---------------------------------------------------------------------------
__global__ __launch_bounds__(256) void swizzle_w_kernel(
    const float* __restrict__ W, unsigned short* __restrict__ Wsw) {
  int tid  = blockIdx.x * 256 + threadIdx.x;    // 0 .. 524287
  int e    = tid & 7;
  int lane = (tid >> 3) & 31;
  int h    = (tid >> 8) & 1;
  int ntg  = (tid >> 9) & 31;
  int kt   = tid >> 14;
  int i = h * 8 + e;
  int k, n;
  if (lane < 16) { k = kt * 32 + i;      n = ntg * 16 + lane;      }
  else           { k = kt * 32 + 16 + i; n = ntg * 16 + lane - 16; }
  Wsw[tid] = f32_to_bf16(W[(size_t)k * A_ + n]);
}

// ---------------------------------------------------------------------------
// Kernel 2: dproj[b][n] = b_attn[n] + sum_k dec[b][k] * W_dec[k][n]
// ---------------------------------------------------------------------------
__global__ __launch_bounds__(512) void decproj_kernel(
    const float* __restrict__ dec, const float* __restrict__ W,
    const float* __restrict__ bias, float* __restrict__ dproj) {
  int b = blockIdx.x;
  int n = threadIdx.x;                   // 512 threads = A
  const float* wd = W + (size_t)EH2_ * A_;
  const float* dh = dec + (size_t)b * DH_;
  float acc = bias[n];
  for (int k = 0; k < DH_; ++k)
    acc += dh[k] * wd[(size_t)k * A_ + n];
  dproj[(size_t)b * A_ + n] = acc;
}

// ---------------------------------------------------------------------------
// Kernel 3: fused  scores[b][s] = sum_n v[n]*tanh( enc[b,s,:]@W_enc[:,n]
//                                                  + dproj[b][n] )
// Block: 512 threads = 16 waves, wave (mi = w>>2, nq = w&3).
// Double-buffered LDS, one barrier per K-step:
//   publish buf[cur] (async_wait + barrier) -> async-fill buf[1-cur] -> compute
// B slice (32KB bf16/step) staged once per block via GLOBAL_LOAD_ASYNC_TO_LDS;
// A tile (64x32) converted f32->bf16 at staging time.
// ---------------------------------------------------------------------------
__global__ __launch_bounds__(512) void scores_kernel(
    const float* __restrict__ enc, const unsigned short* __restrict__ Wsw,
    const float* __restrict__ dproj, const float* __restrict__ vvec,
    float* __restrict__ scores) {
  __shared__ unsigned short Alds[2][MT * KT];     // 2 x 4 KB
  __shared__ unsigned short Blds[2][KT * A_];     // 2 x 32 KB
  __shared__ float sred[4][MT];                   // 1 KB

  const int b   = blockIdx.y;
  const int s0  = blockIdx.x * MT;
  const int tid = threadIdx.x;
  const int lane = tid & 31;
  const int w    = tid >> 5;
  const int mi = w >> 2;       // 0..3 : 16-row M subtile
  const int nq = w & 3;        // 0..3 : 128-col N quarter

  v8f acc[8] = {};

  const size_t encBase = ((size_t)b * S_ + s0) * EH2_;
  const int ar = tid >> 3;           // cooperative A-load row 0..63
  const int aj = (tid & 7) * 4;      // 4 halfs per thread

  // A-fragment addressing for 16-bit A 16x32, contiguous k order (same as B)
  const int arow  = mi * 16 + (lane & 15);
  const int akoff = (lane >> 4) * 16;  // lanes>=16 hold K=16..31 (contiguous)

  // ---- staging helpers ----
  auto fillA = [&](int buf, int kt) {
    float4 f = *(const float4*)(enc + encBase + (size_t)ar * EH2_ + kt * KT + aj);
    unsigned int p0 = (unsigned)f32_to_bf16(f.x) | ((unsigned)f32_to_bf16(f.y) << 16);
    unsigned int p1 = (unsigned)f32_to_bf16(f.z) | ((unsigned)f32_to_bf16(f.w) << 16);
    *(uint2*)&Alds[buf][ar * KT + aj] = make_uint2(p0, p1);
  };
  auto fillB = [&](int buf, int kt) {
    const unsigned short* gp = Wsw + (size_t)kt * (KT * A_);
#pragma unroll
    for (int i = 0; i < 4; ++i) {
      int chunk = tid + i * 512;          // 2048 x 16B = 32 KB
#if HAS_ASYNC_LDS
      __builtin_amdgcn_global_load_async_to_lds_b128(
          (GV4*)(gp + chunk * 8), (LV4*)&Blds[buf][chunk * 8], 0, 0);
#else
      uint4 d = *(const uint4*)(gp + chunk * 8);
      *(uint4*)&Blds[buf][chunk * 8] = d;
#endif
    }
  };

  // prologue: fill buffer 0
  fillA(0, 0);
  fillB(0, 0);

  for (int kt = 0; kt < NKT; ++kt) {
    const int cur = kt & 1;
    // publish buf[cur]: my async fills done; barrier makes all fills visible
    // (also guarantees every wave finished reading buf[1-cur] last step, since
    //  fragment ds_loads are dscnt-waited before the WMMAs that precede it)
    async_wait_all();
    __syncthreads();
    if (kt + 1 < NKT) {          // overlap next-step fill with this compute
      fillA(1 - cur, kt + 1);
      fillB(1 - cur, kt + 1);
    }

    // ---- A fragment: two 16-byte LDS reads per lane ----
    ABfrag af;
    af.q[0] = *(const uint4*)&Alds[cur][arow * KT + akoff];
    af.q[1] = *(const uint4*)&Alds[cur][arow * KT + akoff + 8];

    // ---- 8 N-tiles: B fragments from LDS (staged once per block) ----
#pragma unroll
    for (int nt = 0; nt < 8; ++nt) {
      int ntg = nq * 8 + nt;
      const unsigned short* bp = &Blds[cur][(ntg * 64 + lane) * 8];
      ABfrag bfr;
      bfr.q[0] = *(const uint4*)bp;            // h = 0 (K 0..7 / 16..23)
      bfr.q[1] = *(const uint4*)(bp + 256);    // h = 1 (K 8..15 / 24..31)
      acc[nt] = __builtin_amdgcn_wmma_f32_16x16x32_bf16(
          false, af.bf, false, bfr.bf, (short)0, acc[nt], false, false);
    }
  }

  // ---- fused epilogue: tanh + dot with v ----
  float dpb[8], vv[8];
  const int ncol = nq * 128 + (lane & 15);
#pragma unroll
  for (int nt = 0; nt < 8; ++nt) {
    int n = ncol + nt * 16;
    dpb[nt] = dproj[(size_t)b * A_ + n];
    vv[nt]  = vvec[n];
  }
  float sc[8];
#pragma unroll
  for (int j = 0; j < 8; ++j) {        // j -> row mi*16 + j (+8 for hi lanes)
    float t = 0.f;
#pragma unroll
    for (int nt = 0; nt < 8; ++nt)
      t += vv[nt] * tanhf(acc[nt][j] + dpb[nt]);
    sc[j] = t;
  }
  // reduce over the 16 lanes of each half (columns of the tile)
#pragma unroll
  for (int j = 0; j < 8; ++j) {
    float t = sc[j];
    for (int off = 1; off < 16; off <<= 1)
      t += __shfl_xor(t, off, 32);
    sc[j] = t;
  }
  if ((lane & 15) == 0) {
    int rbase = mi * 16 + (lane >> 4) * 8;
#pragma unroll
    for (int j = 0; j < 8; ++j)
      sred[nq][rbase + j] = sc[j];     // unique slot per (nq,row): deterministic
  }
  __syncthreads();
  if (tid < MT) {
    float t = sred[0][tid] + sred[1][tid] + sred[2][tid] + sred[3][tid];
    scores[(size_t)b * S_ + s0 + tid] = t;
  }
}

// ---------------------------------------------------------------------------
// Kernel 4: masked softmax over S per batch (matches reference -1e10 mask)
// ---------------------------------------------------------------------------
__global__ __launch_bounds__(256) void softmax_kernel(
    const float* __restrict__ scores, const int* __restrict__ mask,
    float* __restrict__ attnw) {
  __shared__ float redmax[8];
  __shared__ float redsum[8];
  const int b = blockIdx.x;
  const int tid = threadIdx.x;
  float vals[4];
  float m = -3.402823466e38f;
#pragma unroll
  for (int i = 0; i < 4; ++i) {
    int s = tid + i * 256;
    float x = scores[(size_t)b * S_ + s];
    if (mask[(size_t)b * S_ + s] == 0) x = -1.0e10f;
    vals[i] = x;
    m = fmaxf(m, x);
  }
  for (int off = 16; off >= 1; off >>= 1) m = fmaxf(m, __shfl_xor(m, off, 32));
  if ((tid & 31) == 0) redmax[tid >> 5] = m;
  __syncthreads();
  float bm = redmax[0];
#pragma unroll
  for (int i = 1; i < 8; ++i) bm = fmaxf(bm, redmax[i]);

  float ssum = 0.f;
#pragma unroll
  for (int i = 0; i < 4; ++i) {
    vals[i] = __expf(vals[i] - bm);
    ssum += vals[i];
  }
  for (int off = 16; off >= 1; off >>= 1) ssum += __shfl_xor(ssum, off, 32);
  if ((tid & 31) == 0) redsum[tid >> 5] = ssum;
  __syncthreads();
  float tot = 0.f;
#pragma unroll
  for (int i = 0; i < 8; ++i) tot += redsum[i];
  float inv = 1.0f / tot;
#pragma unroll
  for (int i = 0; i < 4; ++i)
    attnw[(size_t)b * S_ + tid + i * 256] = vals[i] * inv;
}

// ---------------------------------------------------------------------------
// Kernel 5: context[b][e] = sum_s w[b][s] * enc[b][s][e]  (bandwidth-bound)
// ---------------------------------------------------------------------------
__global__ __launch_bounds__(256) void context_kernel(
    const float* __restrict__ enc, const float* __restrict__ attnw,
    float* __restrict__ ctx) {
  const int b = blockIdx.y;
  const int e = blockIdx.x * 256 + threadIdx.x;
  const float* ep = enc + ((size_t)b * S_) * EH2_ + e;
  const float* wp = attnw + (size_t)b * S_;
  float a0 = 0.f, a1 = 0.f, a2 = 0.f, a3 = 0.f;
  for (int s = 0; s < S_; s += 4) {
    a0 += wp[s + 0] * ep[(size_t)(s + 0) * EH2_];
    a1 += wp[s + 1] * ep[(size_t)(s + 1) * EH2_];
    a2 += wp[s + 2] * ep[(size_t)(s + 2) * EH2_];
    a3 += wp[s + 3] * ep[(size_t)(s + 3) * EH2_];
  }
  ctx[(size_t)b * EH2_ + e] = (a0 + a1) + (a2 + a3);
}

// ---------------------------------------------------------------------------
extern "C" void kernel_launch(void* const* d_in, const int* in_sizes, int n_in,
                              void* d_out, int out_size, void* d_ws, size_t ws_size,
                              hipStream_t stream) {
  const float* enc  = (const float*)d_in[0];   // (B,S,EH2) f32
  const float* dec  = (const float*)d_in[1];   // (B,DH)    f32
  const int*   mask = (const int*)d_in[2];     // (B,S)     i32
  const float* W    = (const float*)d_in[3];   // (EH2+DH,A) f32
  const float* bias = (const float*)d_in[4];   // (A,)      f32
  const float* vvec = (const float*)d_in[5];   // (A,)      f32

  float* ctx   = (float*)d_out;                          // (B,EH2)
  float* attnw = (float*)d_out + (size_t)B_ * EH2_;      // (B,S)

  // Workspace layout (~1.75 MB total):
  unsigned short* Wsw = (unsigned short*)d_ws;                       // 1 MB bf16
  float* dproj  = (float*)((char*)d_ws + (size_t)EH2_ * A_ * 2);     // 256 KB
  float* scores = dproj + (size_t)B_ * A_;                           // 512 KB

  swizzle_w_kernel<<<(EH2_ * A_) / 256, 256, 0, stream>>>(W, Wsw);
  decproj_kernel<<<B_, A_, 0, stream>>>(dec, W, bias, dproj);

  dim3 gScores(S_ / MT, B_);
  scores_kernel<<<gScores, 512, 0, stream>>>(enc, Wsw, dproj, vvec, scores);

  softmax_kernel<<<B_, 256, 0, stream>>>(scores, mask, attnw);

  dim3 gCtx(EH2_ / 256, B_);
  context_kernel<<<gCtx, 256, 0, stream>>>(enc, attnw, ctx);
}
